// MultiQueryAttentionLayerV2_28020366639417
// MI455X (gfx1250) — compile-verified, matrix operand order checked by reading, not run on hardware
//
#include <hip/hip_runtime.h>
#include <hip/hip_bf16.h>

// MultiQueryAttentionLayerV2 for MI455X (gfx1250), bf16 WMMA pipeline.
// Shapes fixed per reference: B=8, D=256, N=M=1024, H=8, K=V=64.

#define BATCH 8
#define DDIM  256
#define NDIM  1024
#define MDIM  1024
#define HEADS 8
#define KDIM  64
#define VDIM  64

typedef __attribute__((ext_vector_type(16))) __bf16 v16bf;
typedef __attribute__((ext_vector_type(8)))  __bf16 v8bf;
typedef __attribute__((ext_vector_type(8)))  float  v8f;

// ---------------------------------------------------------------------------
// Fragment helpers.
// 16-bit A operand (16x32, MxK): lane L (L<16) holds row M=L, K=0..7 (v[0..7])
// and K=16..23 (v[8..15]); lane L+16 holds row M=L, K=8..15 and K=24..31.
// B operand (32x16, KxN) is symmetric with N in place of M.  Both reduce to
// "two contiguous 16-byte loads from a row with a half-dependent offset".
// ---------------------------------------------------------------------------
__device__ __forceinline__ v16bf frag_row(const __bf16* row, int c0, int koff) {
  v8bf lo = *reinterpret_cast<const v8bf*>(row + c0 + koff);
  v8bf hi = *reinterpret_cast<const v8bf*>(row + c0 + 16 + koff);
  v16bf r;
#pragma unroll
  for (int i = 0; i < 8; ++i) { r[i] = lo[i]; r[i + 8] = hi[i]; }
  return r;
}

__device__ __forceinline__ v8f wmma_bf16(v16bf a, v16bf b, v8f c) {
  return __builtin_amdgcn_wmma_f32_16x16x32_bf16(
      /*neg_a=*/false, a, /*neg_b=*/false, b,
      /*c_mod=*/(short)0, c, /*reuse_a=*/false, /*reuse_b=*/false);
}

// ---------------------------------------------------------------------------
// Kernel 1: f32 [b][D][N] -> bf16 [b][N][D] tiled transpose through LDS.
// grid (N/32, D/32, B), block (32, 8).
// ---------------------------------------------------------------------------
__global__ __launch_bounds__(256) void mqa_transpose_bf16(const float* __restrict__ in,
                                                          __bf16* __restrict__ out) {
  __shared__ float tile[32][33];
  const int b  = blockIdx.z;
  const int n0 = blockIdx.x * 32;
  const int d0 = blockIdx.y * 32;
  const int tx = threadIdx.x, ty = threadIdx.y;
  const float* src = in + (size_t)b * DDIM * NDIM;
#pragma unroll
  for (int j = 0; j < 4; ++j)
    tile[ty + j * 8][tx] = src[(size_t)(d0 + ty + j * 8) * NDIM + n0 + tx];
  __syncthreads();
  __bf16* dst = out + (size_t)b * NDIM * DDIM;
#pragma unroll
  for (int j = 0; j < 4; ++j)
    dst[(size_t)(n0 + ty + j * 8) * DDIM + d0 + tx] = (__bf16)tile[tx][ty + j * 8];
}

// ---------------------------------------------------------------------------
// Kernel 2: weight prep.
//   WqT[h][k][d] <- Wq[h][d][k]   (bf16)
//   WkT[k][d]    <- Wk[d][k]      (bf16)
//   WvT[v][d]    <- Wv[d][v]      (bf16)
//   WoB[d*512+c] <- Wo flat       (bf16, c = v*8+h natural order)
// grid 512 x 256 threads.
// ---------------------------------------------------------------------------
__global__ __launch_bounds__(256) void mqa_prep_weights(
    const float* __restrict__ Wq, const float* __restrict__ Wk,
    const float* __restrict__ Wv, const float* __restrict__ Wo,
    __bf16* __restrict__ WqT, __bf16* __restrict__ WkT,
    __bf16* __restrict__ WvT, __bf16* __restrict__ WoB) {
  const int t = blockIdx.x * 256 + threadIdx.x;
  if (t < HEADS * KDIM * DDIM) {
    int d = t % DDIM, k = (t / DDIM) % KDIM, h = t / (DDIM * KDIM);
    WqT[t] = (__bf16)Wq[((size_t)h * DDIM + d) * KDIM + k];
  }
  if (t < KDIM * DDIM) {
    int d = t % DDIM, k = t / DDIM;
    WkT[t] = (__bf16)Wk[(size_t)d * KDIM + k];
    WvT[t] = (__bf16)Wv[(size_t)d * VDIM + k];
  }
  if (t < DDIM * VDIM * HEADS) WoB[t] = (__bf16)Wo[t];
}

// ---------------------------------------------------------------------------
// Kernel 3: Q projection.  qT[b][h][n][k] = sum_d xT[b][n][d] * Wq[h][d][k].
// Structural ping-pong over the d-contraction (step 64, two buffer sets):
// loads for the next 32-chunk are issued before the WMMAs of the current one
// and land directly in their final registers (no rotation movs).
// grid (N/64, H, B), block 128 (4 waves).
// ---------------------------------------------------------------------------
__global__ __launch_bounds__(128) void mqa_proj_q(const __bf16* __restrict__ xT,
                                                  const __bf16* __restrict__ WqT,
                                                  __bf16* __restrict__ qT) {
  const int b = blockIdx.z, h = blockIdx.y;
  const int wave = threadIdx.x >> 5, lane = threadIdx.x & 31;
  const int l16 = lane & 15, koff = (lane >> 4) << 3;
  const int n0 = blockIdx.x * 64 + wave * 16;
  const __bf16* arow = xT + (size_t)b * NDIM * DDIM + (size_t)(n0 + l16) * DDIM;
  const __bf16* wq   = WqT + (size_t)h * KDIM * DDIM;
  v8f acc[4] = {};
  v16bf aX = frag_row(arow, 0, koff);
  v16bf bX[4];
#pragma unroll
  for (int t = 0; t < 4; ++t) bX[t] = frag_row(wq + (size_t)(t * 16 + l16) * DDIM, 0, koff);
  for (int dc = 0; dc < DDIM; dc += 64) {
    v16bf aY = frag_row(arow, dc + 32, koff);
    v16bf bY[4];
#pragma unroll
    for (int t = 0; t < 4; ++t)
      bY[t] = frag_row(wq + (size_t)(t * 16 + l16) * DDIM, dc + 32, koff);
#pragma unroll
    for (int t = 0; t < 4; ++t) acc[t] = wmma_bf16(aX, bX[t], acc[t]);
    const int dn = (dc + 64 < DDIM) ? dc + 64 : 0;   // branchless tail reload
    aX = frag_row(arow, dn, koff);
#pragma unroll
    for (int t = 0; t < 4; ++t)
      bX[t] = frag_row(wq + (size_t)(t * 16 + l16) * DDIM, dn, koff);
#pragma unroll
    for (int t = 0; t < 4; ++t) acc[t] = wmma_bf16(aY, bY[t], acc[t]);
  }
  __bf16* outb = qT + (size_t)(b * HEADS + h) * NDIM * KDIM;
#pragma unroll
  for (int t = 0; t < 4; ++t)
#pragma unroll
    for (int r = 0; r < 8; ++r)
      outb[(size_t)(n0 + r + koff) * KDIM + t * 16 + l16] = (__bf16)acc[t][r];
}

// ---------------------------------------------------------------------------
// Kernel 4: K projection.  kT[b][m][k] = sum_d vT[b][m][d] * Wk[d][k].
// grid (M/64, B), block 128.  Same ping-pong as proj_q.
// ---------------------------------------------------------------------------
__global__ __launch_bounds__(128) void mqa_proj_k(const __bf16* __restrict__ vT,
                                                  const __bf16* __restrict__ WkT,
                                                  __bf16* __restrict__ kT) {
  const int b = blockIdx.y;
  const int wave = threadIdx.x >> 5, lane = threadIdx.x & 31;
  const int l16 = lane & 15, koff = (lane >> 4) << 3;
  const int m0 = blockIdx.x * 64 + wave * 16;
  const __bf16* arow = vT + (size_t)b * MDIM * DDIM + (size_t)(m0 + l16) * DDIM;
  v8f acc[4] = {};
  v16bf aX = frag_row(arow, 0, koff);
  v16bf bX[4];
#pragma unroll
  for (int t = 0; t < 4; ++t) bX[t] = frag_row(WkT + (size_t)(t * 16 + l16) * DDIM, 0, koff);
  for (int dc = 0; dc < DDIM; dc += 64) {
    v16bf aY = frag_row(arow, dc + 32, koff);
    v16bf bY[4];
#pragma unroll
    for (int t = 0; t < 4; ++t)
      bY[t] = frag_row(WkT + (size_t)(t * 16 + l16) * DDIM, dc + 32, koff);
#pragma unroll
    for (int t = 0; t < 4; ++t) acc[t] = wmma_bf16(aX, bX[t], acc[t]);
    const int dn = (dc + 64 < DDIM) ? dc + 64 : 0;
    aX = frag_row(arow, dn, koff);
#pragma unroll
    for (int t = 0; t < 4; ++t)
      bX[t] = frag_row(WkT + (size_t)(t * 16 + l16) * DDIM, dn, koff);
#pragma unroll
    for (int t = 0; t < 4; ++t) acc[t] = wmma_bf16(aY, bY[t], acc[t]);
  }
  __bf16* outb = kT + (size_t)b * MDIM * KDIM;
#pragma unroll
  for (int t = 0; t < 4; ++t)
#pragma unroll
    for (int r = 0; r < 8; ++r)
      outb[(size_t)(m0 + r + koff) * KDIM + t * 16 + l16] = (__bf16)acc[t][r];
}

// ---------------------------------------------------------------------------
// Kernel 5: V projection.  vmat[b][v][m] = sum_d Wv[d][v] * value[b][d][m].
// Output kept [v][m] so the attention kernel's A fragments are contiguous.
// grid (M/64, B), block 128.  Same ping-pong (A/B roles swapped).
// ---------------------------------------------------------------------------
__global__ __launch_bounds__(128) void mqa_proj_v(const __bf16* __restrict__ vT,
                                                  const __bf16* __restrict__ WvT,
                                                  __bf16* __restrict__ vmat) {
  const int b = blockIdx.y;
  const int wave = threadIdx.x >> 5, lane = threadIdx.x & 31;
  const int l16 = lane & 15, koff = (lane >> 4) << 3;
  const int m0 = blockIdx.x * 64 + wave * 16;
  const __bf16* brow = vT + (size_t)b * MDIM * DDIM + (size_t)(m0 + l16) * DDIM;
  v8f acc[4] = {};
  v16bf bX = frag_row(brow, 0, koff);
  v16bf aX[4];
#pragma unroll
  for (int t = 0; t < 4; ++t) aX[t] = frag_row(WvT + (size_t)(t * 16 + l16) * DDIM, 0, koff);
  for (int dc = 0; dc < DDIM; dc += 64) {
    v16bf bY = frag_row(brow, dc + 32, koff);
    v16bf aY[4];
#pragma unroll
    for (int t = 0; t < 4; ++t)
      aY[t] = frag_row(WvT + (size_t)(t * 16 + l16) * DDIM, dc + 32, koff);
#pragma unroll
    for (int t = 0; t < 4; ++t) acc[t] = wmma_bf16(aX[t], bX, acc[t]);
    const int dn = (dc + 64 < DDIM) ? dc + 64 : 0;
    bX = frag_row(brow, dn, koff);
#pragma unroll
    for (int t = 0; t < 4; ++t)
      aX[t] = frag_row(WvT + (size_t)(t * 16 + l16) * DDIM, dn, koff);
#pragma unroll
    for (int t = 0; t < 4; ++t) acc[t] = wmma_bf16(aY[t], bY, acc[t]);
  }
  __bf16* outb = vmat + (size_t)b * VDIM * MDIM;
#pragma unroll
  for (int t = 0; t < 4; ++t)
#pragma unroll
    for (int r = 0; r < 8; ++r)
      outb[(size_t)(t * 16 + r + koff) * MDIM + m0 + l16] = (__bf16)acc[t][r];
}

// ---------------------------------------------------------------------------
// Kernel 6: fused attention + output projection.
// Block = (b, 16-wide n tile), 4 waves.  Per 32-wide m chunk:
//   phase1: waves split the 8 heads; logit tiles L[m,n] = kT . q   (WMMA)
//   phase2: per-position softmax over the 8 heads (lane-local math);
//           next chunk's k/v fragments are issued between the barriers so
//           they overlap phase-2 VALU and phase-3 WMMAs
//   phase3: wave w accumulates o[v in 16w..16w+15][n] over all heads (WMMA)
// The m loop is structurally ping-ponged (step 64, two fragment sets) so
// the software pipeline needs no register-rotation movs.
// Epilogue: o staged in LDS as B operand over c = v*8+h (512 contraction),
//           res[d][n] = WoB . o   (WMMA), f32 stores to d_out.
// ---------------------------------------------------------------------------
struct KF { v16bf a00, a01, a10, a11, v; };

__global__ __launch_bounds__(128) void mqa_attn(const __bf16* __restrict__ qT,
                                                const __bf16* __restrict__ kT,
                                                const __bf16* __restrict__ vmat,
                                                const __bf16* __restrict__ WoB,
                                                float* __restrict__ out) {
  __shared__ float logitsLDS[512 * 8];   // [(m*16+n)][h]  16 KB
  __shared__ v16bf attnBv[8 * 32];       // [h][lane] B operands, 8 KB
  __shared__ v16bf oBv[16 * 32];         // [cchunk][lane] B operands, 16 KB

  const int b  = blockIdx.y;
  const int n0 = blockIdx.x * 16;
  const int tid = threadIdx.x;
  const int wave = tid >> 5, lane = tid & 31;
  const int l16 = lane & 15, koff = (lane >> 4) << 3;

  const __bf16* kTb = kT + (size_t)b * MDIM * KDIM;

  // Hoist q B-operands for this wave's two heads (constant over the m loop).
  v16bf qB[2][2];
#pragma unroll
  for (int hh = 0; hh < 2; ++hh) {
    const int h = wave * 2 + hh;
    const __bf16* qrow = qT + ((size_t)(b * HEADS + h) * NDIM + n0 + l16) * KDIM;
    __builtin_prefetch(qrow, 0, 3);
    qB[hh][0] = frag_row(qrow, 0, koff);
    qB[hh][1] = frag_row(qrow, 32, koff);
  }

  const int v0 = wave * 16;
  const __bf16* vrow = vmat + (size_t)b * VDIM * MDIM + (size_t)(v0 + l16) * MDIM;
  v8f oacc[8] = {};

  auto load_kv = [&](KF& f, int mc) {
    const __bf16* k0 = kTb + (size_t)(mc + l16) * KDIM;
    const __bf16* k1 = kTb + (size_t)(mc + 16 + l16) * KDIM;
    f.a00 = frag_row(k0, 0, koff);
    f.a01 = frag_row(k0, 32, koff);
    f.a10 = frag_row(k1, 0, koff);
    f.a11 = frag_row(k1, 32, koff);
    f.v   = frag_row(vrow, mc, koff);
  };

  // One full pipeline stage: logits(cur) -> softmax -> o += v.attn(cur),
  // issuing the loads for `nxt` between the two barriers.
  auto attn_step = [&](KF& cur, KF& nxt, int mc_next) {
    // ---- phase 1: logits for this wave's two heads ----
#pragma unroll
    for (int hh = 0; hh < 2; ++hh) {
      const int h = wave * 2 + hh;
#pragma unroll
      for (int s = 0; s < 2; ++s) {
        v8f lacc = {};
        lacc = wmma_bf16(s ? cur.a10 : cur.a00, qB[hh][0], lacc);
        lacc = wmma_bf16(s ? cur.a11 : cur.a01, qB[hh][1], lacc);
#pragma unroll
        for (int r = 0; r < 8; ++r) {
          const int ml = s * 16 + r + koff;          // local m row of C tile
          logitsLDS[(ml * 16 + l16) * 8 + h] = lacc[r];
        }
      }
    }
    __syncthreads();

    // ---- issue next chunk's k/v fragment loads (overlap with softmax) ----
    load_kv(nxt, mc_next);

    // ---- phase 2: softmax over heads, write attn in B-operand layout ----
    __bf16* ab = reinterpret_cast<__bf16*>(attnBv);
#pragma unroll
    for (int p4 = 0; p4 < 4; ++p4) {
      const int p = tid + p4 * 128;                  // p = m_local*16 + n_local
      float l[8];
#pragma unroll
      for (int h = 0; h < 8; ++h) l[h] = logitsLDS[p * 8 + h];
      float mx = l[0];
#pragma unroll
      for (int h = 1; h < 8; ++h) mx = fmaxf(mx, l[h]);
      float e[8], ssum = 0.f;
#pragma unroll
      for (int h = 0; h < 8; ++h) { e[h] = __expf((l[h] - mx) * 0.125f); ssum += e[h]; }
      const float inv = 1.0f / ssum;
      const int ml = p >> 4, nl = p & 15;
      const int dl = nl + (((ml >> 3) & 1) << 4);     // destination lane
      const int slot = (ml & 7) + ((ml >> 4) << 3);   // slot inside lane's v16bf
#pragma unroll
      for (int h = 0; h < 8; ++h)
        ab[(h * 32 + dl) * 16 + slot] = (__bf16)(e[h] * inv);
    }
    __syncthreads();

    // ---- phase 3: o[v-tile][n] += v . attn over all 8 heads ----
#pragma unroll
    for (int h = 0; h < 8; ++h)
      oacc[h] = wmma_bf16(cur.v, attnBv[h * 32 + lane], oacc[h]);
    // attnBv reuse by the next phase 2 is fenced by the barrier after the
    // next phase 1; logitsLDS reuse is fenced by the barrier above.
  };

  KF fX, fY;
  load_kv(fX, 0);
  for (int mc = 0; mc < MDIM; mc += 64) {
    attn_step(fX, fY, mc + 32);                       // chunk mc
    const int mn = (mc + 64 < MDIM) ? mc + 64 : 0;    // branchless tail reload
    attn_step(fY, fX, mn);                            // chunk mc + 32
  }

  // ---- epilogue: stage o as B operand over c = v*8 + h ----
  __bf16* ob = reinterpret_cast<__bf16*>(oBv);
#pragma unroll
  for (int h = 0; h < 8; ++h)
#pragma unroll
    for (int r = 0; r < 8; ++r) {
      const int v = v0 + r + koff;
      const int c = v * 8 + h;
      const int cc = c >> 5, local = c & 31;
      const int dl = l16 + (((local >> 3) & 1) << 4);
      const int slot = (local & 7) + ((local >> 4) << 3);
      ob[(cc * 32 + dl) * 16 + slot] = (__bf16)oacc[h][r];
    }
  __syncthreads();

  // ---- res[d][n-tile] = WoB[d][c] . o[c][n], 4 d-tiles per wave ----
  float* outb = out + (size_t)b * DDIM * NDIM;
#pragma unroll
  for (int t4 = 0; t4 < 4; ++t4) {
    const int t = wave * 4 + t4;
    const __bf16* wrow = WoB + (size_t)(t * 16 + l16) * (VDIM * HEADS);
    v8f racc = {};
#pragma unroll
    for (int cc = 0; cc < 16; ++cc) {
      v16bf aw = frag_row(wrow, cc * 32, koff);
      racc = wmma_bf16(aw, oBv[cc * 32 + lane], racc);
    }
#pragma unroll
    for (int r = 0; r < 8; ++r) {
      const int d = t * 16 + r + koff;
      outb[(size_t)d * NDIM + n0 + l16] = racc[r];
    }
  }
}

// ---------------------------------------------------------------------------
// Host-side launcher.
// ---------------------------------------------------------------------------
extern "C" void kernel_launch(void* const* d_in, const int* in_sizes, int n_in,
                              void* d_out, int out_size, void* d_ws, size_t ws_size,
                              hipStream_t stream) {
  const float* x     = (const float*)d_in[0];  // [8][256][32][32]
  const float* value = (const float*)d_in[1];  // [8][256][32][32]
  const float* Wq    = (const float*)d_in[2];  // [8][256][64]
  const float* Wk    = (const float*)d_in[3];  // [256][64]
  const float* Wv    = (const float*)d_in[4];  // [256][64]
  const float* Wo    = (const float*)d_in[5];  // [256][64][8]
  float* out = (float*)d_out;

  char* w = (char*)d_ws;
  __bf16* xbfT = (__bf16*)w; w += (size_t)BATCH * NDIM * DDIM * 2;          // 4 MB
  __bf16* vbfT = (__bf16*)w; w += (size_t)BATCH * MDIM * DDIM * 2;          // 4 MB
  __bf16* WqT  = (__bf16*)w; w += (size_t)HEADS * KDIM * DDIM * 2;          // 256 KB
  __bf16* WkT  = (__bf16*)w; w += (size_t)KDIM * DDIM * 2;                  // 32 KB
  __bf16* WvT  = (__bf16*)w; w += (size_t)VDIM * DDIM * 2;                  // 32 KB
  __bf16* WoB  = (__bf16*)w; w += (size_t)DDIM * VDIM * HEADS * 2;          // 256 KB
  __bf16* qTb  = (__bf16*)w; w += (size_t)BATCH * HEADS * NDIM * KDIM * 2;  // 8 MB
  __bf16* kTb  = (__bf16*)w; w += (size_t)BATCH * MDIM * KDIM * 2;          // 1 MB
  __bf16* vmat = (__bf16*)w; w += (size_t)BATCH * VDIM * MDIM * 2;          // 1 MB

  dim3 tgrid(NDIM / 32, DDIM / 32, BATCH), tblk(32, 8);
  mqa_transpose_bf16<<<tgrid, tblk, 0, stream>>>(x, xbfT);
  mqa_transpose_bf16<<<tgrid, tblk, 0, stream>>>(value, vbfT);

  mqa_prep_weights<<<dim3(512), dim3(256), 0, stream>>>(Wq, Wk, Wv, Wo,
                                                        WqT, WkT, WvT, WoB);

  mqa_proj_q<<<dim3(NDIM / 64, HEADS, BATCH), dim3(128), 0, stream>>>(xbfT, WqT, qTb);
  mqa_proj_k<<<dim3(MDIM / 64, BATCH), dim3(128), 0, stream>>>(vbfT, WkT, kTb);
  mqa_proj_v<<<dim3(MDIM / 64, BATCH), dim3(128), 0, stream>>>(vbfT, WvT, vmat);

  mqa_attn<<<dim3(NDIM / 16, BATCH), dim3(128), 0, stream>>>(qTb, kTb, vmat, WoB, out);
}
